// RnnInferenceNetwork_58188216926821
// MI455X (gfx1250) — matrix operand order repaired and checked by reading, not compiled, not run on hardware
//
#include <hip/hip_runtime.h>

typedef __bf16 bf16;
typedef __attribute__((ext_vector_type(8)))  bf16  v8bf;
typedef __attribute__((ext_vector_type(16))) bf16  v16bf;
typedef __attribute__((ext_vector_type(8)))  float v8f;
typedef __attribute__((ext_vector_type(4)))  unsigned v4u;
typedef __attribute__((ext_vector_type(4)))  int   v4i;
typedef __attribute__((ext_vector_type(8)))  int   v8i;

#define DEVINL __device__ __forceinline__

// ---------------- problem constants ----------------
constexpr int BB  = 64;    // batch
constexpr int TT  = 1024;  // time steps
constexpr int DD  = 256;   // input dim
constexpr int HH  = 512;   // hidden
constexpr int LLn = 64;    // latent L
constexpr int SSn = 4;     // samples S
constexpr int SBr = 256;   // S*B rows per step
constexpr int DLk = 320;   // D+L (rnn_in K)
constexpr int NWG = 16;    // persistent workgroups
constexpr int HSL = 32;    // H columns owned per WG
constexpr float kLOG2PI = 1.8378770664093453f;

// ---------------- workspace layout (bytes) ----------------
constexpr size_t OFF_WXT = 0;                                   // [1536][320] bf16 (N-major)
constexpr size_t OFF_WHT = OFF_WXT + (size_t)1536*DLk*2;        // [1536][512] bf16
constexpr size_t OFF_WDT = OFF_WHT + (size_t)1536*HH*2;         // [128][512]  bf16
constexpr size_t OFF_XBF = OFF_WDT + (size_t)128*HH*2;          // [T][64][256] bf16
constexpr size_t OFF_HBF = OFF_XBF + (size_t)TT*BB*DD*2;        // [256][512] bf16 state h
constexpr size_t OFF_ZA  = OFF_HBF + (size_t)SBr*HH*2;          // [256][64]  bf16 state z
constexpr size_t OFF_RH  = OFF_ZA  + (size_t)SBr*LLn*2;         // [256][512] bf16 r*h
constexpr size_t OFF_PE  = OFF_RH  + (size_t)SBr*HH*2;          // [4][256] f32 ent partials
constexpr size_t OFF_PL  = OFF_PE  + 4096;                      // [4][256] f32 lp partials
constexpr size_t OFF_CTR = OFF_PL  + 4096;                      // barrier counter
constexpr size_t WS_NEED = OFF_CTR + 256;

// ---------------- LDS layout (bytes) ----------------
constexpr int SM_WX  = 0;                         // [3][32][320] bf16 = 61440
constexpr int SM_WH  = SM_WX + 3*HSL*DLk*2;       // [3][32][512] bf16 = 98304
constexpr int SM_WD  = SM_WH + 3*HSL*HH*2;        // [2][16][512] bf16 = 32768
constexpr int SM_Z   = SM_WD + 2*16*HH*2;         // [256][32] f32 = 32768 (z-gate; reused as dp buf)
constexpr int SM_RED = SM_Z  + SBr*HSL*4;         // [64][4][2] f32 = 2048
constexpr int SM_B   = SM_RED + 64*4*2*4;         // [3][32] f32
constexpr int SM_BD  = SM_B  + 3*HSL*4;           // [2][16] f32
constexpr int SMEM_BYTES = SM_BD + 32*4;          // 227840 B  (< 320KB WGP LDS)

// ---------------- helpers ----------------
union V16U { v16bf v; v8bf h[2]; };

DEVINL v16bf load16(const bf16* p0, const bf16* p1) {
  V16U u; u.h[0] = *(const v8bf*)p0; u.h[1] = *(const v8bf*)p1; return u.v;
}
DEVINL v8f wmma_bf16(v16bf a, v16bf b, v8f c) {
  // D(16x16 f32) = A(16x32 bf16) * B(32x16 bf16) + C
  return __builtin_amdgcn_wmma_f32_16x16x32_bf16(false, a, false, b, (short)0, c, false, false);
}
DEVINL float sigmoidf_(float x) { return 1.0f / (1.0f + __expf(-x)); }
DEVINL float softplusf_(float x) { return x > 20.0f ? x : log1pf(__expf(x)); }

// device-scope counting barrier across the 16 persistent WGs
DEVINL void gbarrier(unsigned* ctr, unsigned& num) {
  __syncthreads();
  if (threadIdx.x == 0) {
    __threadfence();
    ++num;
    __hip_atomic_fetch_add(ctr, 1u, __ATOMIC_RELEASE, __HIP_MEMORY_SCOPE_AGENT);
    while (__hip_atomic_load(ctr, __ATOMIC_ACQUIRE, __HIP_MEMORY_SCOPE_AGENT) < num * NWG) {
      __builtin_amdgcn_s_sleep(1);
    }
  }
  __syncthreads();
  __threadfence();
}

// Tensor Data Mover: DMA a 2-D bf16 tile (rows x row_elems, global row stride
// row_stride_elems) into LDS at byte offset lds_off.  D# per CDNA5 ISA ch.8:
//   group0: count=1 | lds_addr | global_addr[56:0] | type=2
//   group1: data_size=2B, tensor_dim0/1, tile_dim0/1, tensor_dim0_stride
DEVINL void tdm_load_2d(unsigned lds_off, const void* gaddr,
                        unsigned rows, unsigned row_elems, unsigned row_stride_elems)
{
  unsigned long long ga = (unsigned long long)(uintptr_t)gaddr;
  v4u g0;
  g0.x = 1u;                                                   // count=1 (user D#)
  g0.y = lds_off;                                              // LDS byte address
  g0.z = (unsigned)ga;                                         // global addr [31:0]
  g0.w = ((unsigned)(ga >> 32) & 0x01FFFFFFu) | 0x80000000u;   // addr[56:32] | type=2
  v8i g1;
  g1[0] = 0x00010000;                                          // data_size=1 -> 2 bytes/elem
  g1[1] = (int)((row_elems & 0xFFFFu) << 16);                  // tensor_dim0[15:0]
  g1[2] = (int)(((row_elems >> 16) & 0xFFFFu) |                // tensor_dim0[31:16]
                ((rows & 0xFFFFu) << 16));                     // tensor_dim1[15:0]
  g1[3] = (int)(((rows >> 16) & 0xFFFFu) |                     // tensor_dim1[31:16]
                ((row_elems & 0xFFFFu) << 16));                // tile_dim0
  g1[4] = (int)(rows & 0xFFFFu);                               // tile_dim1 (tile_dim2=0)
  g1[5] = (int)row_stride_elems;                               // tensor_dim0_stride[31:0]
  g1[6] = 0;                                                   // stride0 hi / stride1 lo
  g1[7] = 0;
  v4i z4 = {0, 0, 0, 0};
  v8i z8 = {0, 0, 0, 0, 0, 0, 0, 0};
  __builtin_amdgcn_tensor_load_to_lds(g0, g1, z4, z4, z8, 0);
}

// ---------------- prep: bf16 transposed weights, bf16 inputs, zero state ----------------
__global__ void prep_kernel(const float* __restrict__ inputs,
                            const float* __restrict__ W_x,
                            const float* __restrict__ W_h,
                            const float* __restrict__ W_d,
                            bf16* __restrict__ WxT, bf16* __restrict__ WhT,
                            bf16* __restrict__ WdT, bf16* __restrict__ xbf,
                            bf16* __restrict__ hbf, bf16* __restrict__ zA,
                            unsigned* __restrict__ ctr)
{
  const size_t tid    = (size_t)blockIdx.x * blockDim.x + threadIdx.x;
  const size_t stride = (size_t)gridDim.x * blockDim.x;
  for (size_t i = tid; i < (size_t)1536*DLk; i += stride) {   // W_x (DL,3H) -> [n][k]
    size_t n = i / DLk, k = i % DLk; WxT[i] = (bf16)W_x[k*1536 + n];
  }
  for (size_t i = tid; i < (size_t)1536*HH; i += stride) {    // W_h (H,3H) -> [n][k]
    size_t n = i / HH, k = i % HH;  WhT[i] = (bf16)W_h[k*1536 + n];
  }
  for (size_t i = tid; i < (size_t)128*HH; i += stride) {     // W_d (H,2L) -> [n][k]
    size_t n = i / HH, k = i % HH;  WdT[i] = (bf16)W_d[k*128 + n];
  }
  for (size_t i = tid; i < (size_t)TT*BB*DD; i += stride) {   // inputs (B,T,D) -> x[t][b][d]
    size_t d = i % DD, b = (i / DD) % BB, t = i / ((size_t)DD*BB);
    xbf[i] = (bf16)inputs[(b*TT + t)*DD + d];
  }
  for (size_t i = tid; i < (size_t)SBr*HH;  i += stride) hbf[i] = (bf16)0.0f;
  for (size_t i = tid; i < (size_t)SBr*LLn; i += stride) zA[i]  = (bf16)0.0f;
  if (tid == 0) *ctr = 0u;
}

// ---------------- persistent GRU scan kernel ----------------
__global__ void __launch_bounds__(256, 1)
rnn_persistent(const float* __restrict__ noise,
               const float* __restrict__ bias,     // b (3H)
               const float* __restrict__ bias_d,   // b_d (2L)
               const bf16* __restrict__ WxT, const bf16* __restrict__ WhT,
               const bf16* __restrict__ WdT, const bf16* __restrict__ xbf,
               bf16* __restrict__ hbf,  bf16* __restrict__ zA,
               bf16* __restrict__ rhb,
               float* __restrict__ part_e, float* __restrict__ part_l,
               unsigned* __restrict__ ctr,
               float* __restrict__ out_z, float* __restrict__ out_ent,
               float* __restrict__ out_lp)
{
  extern __shared__ char smem[];
  bf16*  sWx = (bf16*)(smem + SM_WX);
  bf16*  sWh = (bf16*)(smem + SM_WH);
  bf16*  sWd = (bf16*)(smem + SM_WD);
  float* sZ  = (float*)(smem + SM_Z);    // z-gate [256][32]
  float* sDp = (float*)(smem + SM_Z);    // reuse as dp buf [64][32] in phase 3
  float* sRed= (float*)(smem + SM_RED);
  float* sB  = (float*)(smem + SM_B);
  float* sBd = (float*)(smem + SM_BD);

  const int wg   = blockIdx.x;       // 0..15
  const int tid  = threadIdx.x;
  const int wave = tid >> 5;
  const int lane = tid & 31;
  const int hf   = lane >> 4;        // wave half
  const int n16  = lane & 15;
  const int c0   = wg * HSL;         // owned h-column base
  const int cm   = wg & 3;           // dp latent col-tile
  const int r0   = (wg >> 2) * 64;   // dp row base

  // ---- stage weight slices into LDS via the Tensor Data Mover (one wave issues) ----
  if (wave == 0) {
    const unsigned ldsBase = (unsigned)(uintptr_t)(void*)smem;
    for (int g = 0; g < 3; ++g) {
      tdm_load_2d(ldsBase + SM_WX + g*HSL*DLk*2,
                  WxT + (size_t)(g*HH + c0)*DLk, HSL, DLk, DLk);
      tdm_load_2d(ldsBase + SM_WH + g*HSL*HH*2,
                  WhT + (size_t)(g*HH + c0)*HH,  HSL, HH,  HH);
    }
    for (int d = 0; d < 2; ++d) {
      tdm_load_2d(ldsBase + SM_WD + d*16*HH*2,
                  WdT + (size_t)(d*LLn + cm*16)*HH, 16, HH, HH);
    }
    __builtin_amdgcn_s_wait_tensorcnt(0);
  }
  if (tid < 3*HSL) sB[tid]  = bias[(tid/HSL)*HH + c0 + (tid % HSL)];
  if (tid < 32)    sBd[tid] = bias_d[(tid >> 4)*LLn + cm*16 + (tid & 15)];
  __syncthreads();

  unsigned barNum = 0;
  const v8f vz = {0.f,0.f,0.f,0.f,0.f,0.f,0.f,0.f};

  for (int t = 0; t < TT; ++t) {
    // phase 0: finalize ent/lp of step t-1 (partials synced by last gbarrier)
    if (t > 0 && tid < 16) {
      const int m = wg*16 + tid;
      float e = part_e[m] + part_e[SBr+m] + part_e[2*SBr+m] + part_e[3*SBr+m];
      float l = part_l[m] + part_l[SBr+m] + part_l[2*SBr+m] + part_l[3*SBr+m];
      out_ent[(size_t)m*TT + (t-1)] = e;
      out_lp [(size_t)m*TT + (t-1)] = l;
    }
    const bf16* xt = xbf + (size_t)t*BB*DD;

    // ---------- phase 1: z,r gates + write r*h ----------
    for (int rep = 0; rep < 2; ++rep) {
      const int rt = wave*2 + rep;
      const int mA = rt*16 + n16;
      __builtin_prefetch(hbf + (size_t)mA*HH, 0, 1);
      v8f acc[4] = {vz, vz, vz, vz};          // z ct0, z ct1, r ct0, r ct1
      for (int kt = 0; kt < 10; ++kt) {       // rnn_in = [x | z_prev], K = 320
        const int k0 = kt*32;
        const bf16* ap = (k0 < DD) ? (xt + (size_t)(mA & (BB-1))*DD + k0 + hf*16)
                                   : (zA + (size_t)mA*LLn + (k0 - DD) + hf*16);
        const v16bf a = load16(ap, ap + 8);
        const bf16* pz0 = sWx + (size_t)(0*HSL +      n16)*DLk + k0 + hf*16;
        const bf16* pz1 = sWx + (size_t)(0*HSL + 16 + n16)*DLk + k0 + hf*16;
        const bf16* pr0 = sWx + (size_t)(1*HSL +      n16)*DLk + k0 + hf*16;
        const bf16* pr1 = sWx + (size_t)(1*HSL + 16 + n16)*DLk + k0 + hf*16;
        const v16bf bz0 = load16(pz0, pz0+8), bz1 = load16(pz1, pz1+8);
        const v16bf br0 = load16(pr0, pr0+8), br1 = load16(pr1, pr1+8);
        acc[0] = wmma_bf16(a, bz0, acc[0]);
        acc[1] = wmma_bf16(a, bz1, acc[1]);
        acc[2] = wmma_bf16(a, br0, acc[2]);
        acc[3] = wmma_bf16(a, br1, acc[3]);
      }
      for (int kt = 0; kt < 16; ++kt) {       // h @ W_h[:, :2H], K = 512
        const int k0 = kt*32;
        const bf16* ap = hbf + (size_t)mA*HH + k0 + hf*16;
        const v16bf a = load16(ap, ap + 8);
        const bf16* pz0 = sWh + (size_t)(0*HSL +      n16)*HH + k0 + hf*16;
        const bf16* pz1 = sWh + (size_t)(0*HSL + 16 + n16)*HH + k0 + hf*16;
        const bf16* pr0 = sWh + (size_t)(1*HSL +      n16)*HH + k0 + hf*16;
        const bf16* pr1 = sWh + (size_t)(1*HSL + 16 + n16)*HH + k0 + hf*16;
        const v16bf bz0 = load16(pz0, pz0+8), bz1 = load16(pz1, pz1+8);
        const v16bf br0 = load16(pr0, pr0+8), br1 = load16(pr1, pr1+8);
        acc[0] = wmma_bf16(a, bz0, acc[0]);
        acc[1] = wmma_bf16(a, bz1, acc[1]);
        acc[2] = wmma_bf16(a, br0, acc[2]);
        acc[3] = wmma_bf16(a, br1, acc[3]);
      }
#pragma unroll
      for (int ct = 0; ct < 2; ++ct)
#pragma unroll
        for (int v = 0; v < 8; ++v) {
          const int m  = rt*16 + hf*8 + v;    // C-layout: M = v + 8*half
          const int cl = ct*16 + n16;
          const float zv = sigmoidf_(acc[ct][v]   + sB[cl]);
          const float rv = sigmoidf_(acc[2+ct][v] + sB[HSL + cl]);
          sZ[m*HSL + cl] = zv;
          const float hv = (float)hbf[(size_t)m*HH + c0 + cl];
          rhb[(size_t)m*HH + c0 + cl] = (bf16)(rv * hv);
        }
    }
    gbarrier(ctr, barNum);

    // ---------- phase 2: n = tanh(xg_n + (r*h)@W_h_n); h_new ----------
    for (int rep = 0; rep < 2; ++rep) {
      const int rt = wave*2 + rep;
      const int mA = rt*16 + n16;
      v8f accn[2] = {vz, vz};
      for (int kt = 0; kt < 10; ++kt) {
        const int k0 = kt*32;
        const bf16* ap = (k0 < DD) ? (xt + (size_t)(mA & (BB-1))*DD + k0 + hf*16)
                                   : (zA + (size_t)mA*LLn + (k0 - DD) + hf*16);
        const v16bf a = load16(ap, ap + 8);
        const bf16* pn0 = sWx + (size_t)(2*HSL +      n16)*DLk + k0 + hf*16;
        const bf16* pn1 = sWx + (size_t)(2*HSL + 16 + n16)*DLk + k0 + hf*16;
        const v16bf bn0 = load16(pn0, pn0+8), bn1 = load16(pn1, pn1+8);
        accn[0] = wmma_bf16(a, bn0, accn[0]);
        accn[1] = wmma_bf16(a, bn1, accn[1]);
      }
      for (int kt = 0; kt < 16; ++kt) {
        const int k0 = kt*32;
        const bf16* ap = rhb + (size_t)mA*HH + k0 + hf*16;
        const v16bf a = load16(ap, ap + 8);
        const bf16* pn0 = sWh + (size_t)(2*HSL +      n16)*HH + k0 + hf*16;
        const bf16* pn1 = sWh + (size_t)(2*HSL + 16 + n16)*HH + k0 + hf*16;
        const v16bf bn0 = load16(pn0, pn0+8), bn1 = load16(pn1, pn1+8);
        accn[0] = wmma_bf16(a, bn0, accn[0]);
        accn[1] = wmma_bf16(a, bn1, accn[1]);
      }
#pragma unroll
      for (int ct = 0; ct < 2; ++ct)
#pragma unroll
        for (int v = 0; v < 8; ++v) {
          const int m  = rt*16 + hf*8 + v;
          const int cl = ct*16 + n16;
          const float nv = tanhf(accn[ct][v] + sB[2*HSL + cl]);
          const float zv = sZ[m*HSL + cl];
          const float hv = (float)hbf[(size_t)m*HH + c0 + cl];
          hbf[(size_t)m*HH + c0 + cl] = (bf16)((1.0f - zv)*nv + zv*hv);
        }
    }
    gbarrier(ctr, barNum);

    // ---------- phase 3: dp = h_new @ W_d + b_d ; sample z, ent/lp partials ----------
    {
      const int d   = wave >> 2;               // 0=mu tile, 1=raw tile
      const int rtl = wave & 3;
      const int rt  = (r0 >> 4) + rtl;
      const int mA  = rt*16 + n16;
      v8f acc = vz;
      for (int kt = 0; kt < 16; ++kt) {
        const int k0 = kt*32;
        const bf16* ap = hbf + (size_t)mA*HH + k0 + hf*16;
        const v16bf a = load16(ap, ap + 8);
        const bf16* pw = sWd + (size_t)(d*16 + n16)*HH + k0 + hf*16;
        acc = wmma_bf16(a, load16(pw, pw+8), acc);
      }
#pragma unroll
      for (int v = 0; v < 8; ++v) {
        const int ml = rtl*16 + hf*8 + v;      // local row 0..63
        sDp[ml*32 + d*16 + n16] = acc[v] + sBd[d*16 + n16];
      }
    }
    __syncthreads();
    {
      const int ml = tid & 63;
      const int q  = tid >> 6;                 // quarter of the 16 latent cols
      const int m  = r0 + ml;
      float pe = 0.f, pl = 0.f;
#pragma unroll
      for (int j = 0; j < 4; ++j) {
        const int lc = q*4 + j;                // 0..15 within col-tile
        const int l  = cm*16 + lc;
        const float mu  = sDp[ml*32 + lc];
        const float raw = sDp[ml*32 + 16 + lc];
        const float sg  = softplusf_(raw) + 1e-4f;
        const float ls  = __logf(sg);
        const float ep  = noise[((size_t)t*SBr + m)*LLn + l];
        const float zs  = mu + sg*ep;
        out_z[((size_t)m*TT + t)*LLn + l] = zs; // (S,B,T,L): row m == s*B+b
        zA[(size_t)m*LLn + l] = (bf16)zs;       // feeds next step's rnn_in
        pe += 0.5f*(kLOG2PI + 1.0f) + ls;
        pl += -0.5f*ep*ep - ls - 0.5f*kLOG2PI;
      }
      sRed[(ml*4 + q)*2 + 0] = pe;
      sRed[(ml*4 + q)*2 + 1] = pl;
    }
    __syncthreads();
    if (tid < 64) {
      const int m = r0 + tid;
      float e = 0.f, l = 0.f;
#pragma unroll
      for (int q = 0; q < 4; ++q) { e += sRed[(tid*4+q)*2]; l += sRed[(tid*4+q)*2 + 1]; }
      part_e[cm*SBr + m] = e;
      part_l[cm*SBr + m] = l;
    }
    gbarrier(ctr, barNum);
  }

  // flush last step's ent/lp
  if (tid < 16) {
    const int m = wg*16 + tid;
    float e = part_e[m] + part_e[SBr+m] + part_e[2*SBr+m] + part_e[3*SBr+m];
    float l = part_l[m] + part_l[SBr+m] + part_l[2*SBr+m] + part_l[3*SBr+m];
    out_ent[(size_t)m*TT + (TT-1)] = e;
    out_lp [(size_t)m*TT + (TT-1)] = l;
  }
}

// ---------------- host launcher ----------------
extern "C" void kernel_launch(void* const* d_in, const int* in_sizes, int n_in,
                              void* d_out, int out_size, void* d_ws, size_t ws_size,
                              hipStream_t stream)
{
  if (ws_size < WS_NEED) return;
  const float* inputs = (const float*)d_in[0];
  const float* noise  = (const float*)d_in[1];
  const float* W_x    = (const float*)d_in[2];
  const float* W_h    = (const float*)d_in[3];
  const float* b      = (const float*)d_in[4];
  const float* W_d    = (const float*)d_in[5];
  const float* b_d    = (const float*)d_in[6];

  char* ws = (char*)d_ws;
  bf16* WxT = (bf16*)(ws + OFF_WXT);
  bf16* WhT = (bf16*)(ws + OFF_WHT);
  bf16* WdT = (bf16*)(ws + OFF_WDT);
  bf16* xbf = (bf16*)(ws + OFF_XBF);
  bf16* hbf = (bf16*)(ws + OFF_HBF);
  bf16* zAb = (bf16*)(ws + OFF_ZA);
  bf16* rhb = (bf16*)(ws + OFF_RH);
  float* part_e = (float*)(ws + OFF_PE);
  float* part_l = (float*)(ws + OFF_PL);
  unsigned* ctr = (unsigned*)(ws + OFF_CTR);

  float* out     = (float*)d_out;
  float* out_z   = out;                                  // (S,B,T,L)
  float* out_ent = out + (size_t)SSn*BB*TT*LLn;          // (S,B,T)
  float* out_lp  = out_ent + (size_t)SSn*BB*TT;          // (S,B,T)

  prep_kernel<<<dim3(2048), dim3(256), 0, stream>>>(
      inputs, W_x, W_h, W_d, WxT, WhT, WdT, xbf, hbf, zAb, ctr);

  rnn_persistent<<<dim3(NWG), dim3(256), SMEM_BYTES, stream>>>(
      noise, b, b_d, WxT, WhT, WdT, xbf, hbf, zAb, rhb,
      part_e, part_l, ctr, out_z, out_ent, out_lp);
}
